// HolEInteraction_20615843021596
// MI455X (gfx1250) — matrix operand order repaired and checked by reading, not compiled, not run on hardware
//
#include <hip/hip_runtime.h>

// HolE scoring via frequency-domain identity, f16 WMMA GEMMs against a shared
// rDFT basis (gfx1250 / CDNA5, wave32).
//
//   score_b = (1/D) * sum_k w_k * Re( conj(H_k) * T_k * conj(R_k) )
//   X = xc - i*xs,  xc[k] = sum_i x_i cos(2*pi*i*k/D), xs[k] = sum_i x_i sin(...)
//   Re(...) = (hc*tc + hs*ts)*rc - (hs*tc - hc*ts)*rs
//
// Workspace layout (d_ws): cos basis [272][512] f16, then sin basis [272][512] f16
// (557,056 bytes total). Bins 257..271 are zero padding.

typedef __attribute__((ext_vector_type(16))) _Float16 v16h;
typedef __attribute__((ext_vector_type(8)))  _Float16 v8h;
typedef __attribute__((ext_vector_type(8)))  float    v8f;

#define DD     512
#define NBINS  257            // rfft bins for D=512
#define NTILE  17             // ceil(257/16) column tiles of 16 bins
#define NBPAD  (NTILE * 16)   // 272
#define ROWS   16             // batch rows per workgroup
#define NTHR   256            // 8 waves of 32

union V16 { v16h v; struct { v8h lo, hi; } s; };

// ---------------------------------------------------------------------------
// Basis init: cosb[n][i] = cos(2*pi*n*i/512), sinb[n][i] = sin(...), f16.
// Angle reduced exactly via (n*i) mod 512 before the float conversion.
// ---------------------------------------------------------------------------
__global__ void hole_basis_init(_Float16* __restrict__ cosb,
                                _Float16* __restrict__ sinb) {
    int idx = blockIdx.x * blockDim.x + threadIdx.x;
    if (idx >= NBPAD * DD) return;
    int n = idx >> 9;          // bin index
    int i = idx & (DD - 1);    // sample index
    if (n < NBINS) {
        float ang = 0.012271846303085129f /* 2*pi/512 */ *
                    (float)((n * i) & (DD - 1));
        cosb[idx] = (_Float16)__cosf(ang);
        sinb[idx] = (_Float16)__sinf(ang);
    } else {
        cosb[idx] = (_Float16)0.0f;
        sinb[idx] = (_Float16)0.0f;
    }
}

// ---------------------------------------------------------------------------
// Main kernel: one workgroup = 16 batch rows. Stage h/t/r rows as f16 in LDS,
// 8 waves cover the 17 bin tiles; 6 WMMA chains (K=512) per tile produce the
// six spectra tiles, reduced in-register with Parseval weights.
// ---------------------------------------------------------------------------
__global__ __launch_bounds__(NTHR) void hole_main(
        const float* __restrict__ gh,
        const float* __restrict__ gt,
        const float* __restrict__ gr,
        const _Float16* __restrict__ cosb,
        const _Float16* __restrict__ sinb,
        float* __restrict__ out) {
    __shared__ _Float16 sh[ROWS * DD];
    __shared__ _Float16 st[ROWS * DD];
    __shared__ _Float16 sr[ROWS * DD];
    __shared__ float    part[NTHR / 32][ROWS];

    const int  tid  = threadIdx.x;
    const long row0 = (long)blockIdx.x * ROWS;
    const long g0   = row0 * DD;

    // Stage 16 rows of each input into LDS as f16 (coalesced f32 reads).
    for (int idx = tid; idx < ROWS * DD; idx += NTHR) {
        sh[idx] = (_Float16)gh[g0 + idx];
        st[idx] = (_Float16)gt[g0 + idx];
        sr[idx] = (_Float16)gr[g0 + idx];
    }
    __syncthreads();

    const int wave = tid >> 5;
    const int lane = tid & 31;
    const int m16  = lane & 15;   // A-row / B-col / C-col index within tile
    const int hlf  = lane >> 4;   // which 16-lane half of the wave

    float red[8];
#pragma unroll
    for (int v = 0; v < 8; ++v) red[v] = 0.0f;

    // Wave-uniform tile loop: EXEC stays all-ones for every WMMA.
    for (int tile = wave; tile < NTILE; tile += NTHR / 32) {
        v8f ahc = {}, ahs = {}, atc = {}, ats = {}, arc = {}, ars = {};
        const int n = tile * 16 + m16;          // this lane's bin column

        const _Float16* bcrow = cosb + n * DD;
        const _Float16* bsrow = sinb + n * DD;
        const _Float16* arow_h = sh + m16 * DD;
        const _Float16* arow_t = st + m16 * DD;
        const _Float16* arow_r = sr + m16 * DD;

#pragma unroll 4
        for (int ks = 0; ks < DD / 32; ++ks) {
            // A fragment (16x32 f16): lane holds row m16; halfs e are
            // K = 32ks + 16*hlf + e (contiguous, matching B).
            const int a0 = ks * 32 + hlf * 16;
            V16 ah, at_, ar_;
            ah.s.lo  = *(const v8h*)(arow_h + a0);
            ah.s.hi  = *(const v8h*)(arow_h + a0 + 8);
            at_.s.lo = *(const v8h*)(arow_t + a0);
            at_.s.hi = *(const v8h*)(arow_t + a0 + 8);
            ar_.s.lo = *(const v8h*)(arow_r + a0);
            ar_.s.hi = *(const v8h*)(arow_r + a0 + 8);

            // B fragment (32x16 f16): lane holds column n; halfs e hold
            // K = 32ks + 16*hlf + e  -> 16 contiguous halfs (32B).
            const int i0 = ks * 32 + hlf * 16;
            v16h bc = *(const v16h*)(bcrow + i0);
            v16h bs = *(const v16h*)(bsrow + i0);

            ahc = __builtin_amdgcn_wmma_f32_16x16x32_f16(false, ah.v,  false, bc, (short)0, ahc, false, false);
            ahs = __builtin_amdgcn_wmma_f32_16x16x32_f16(false, ah.v,  false, bs, (short)0, ahs, false, false);
            atc = __builtin_amdgcn_wmma_f32_16x16x32_f16(false, at_.v, false, bc, (short)0, atc, false, false);
            ats = __builtin_amdgcn_wmma_f32_16x16x32_f16(false, at_.v, false, bs, (short)0, ats, false, false);
            arc = __builtin_amdgcn_wmma_f32_16x16x32_f16(false, ar_.v, false, bc, (short)0, arc, false, false);
            ars = __builtin_amdgcn_wmma_f32_16x16x32_f16(false, ar_.v, false, bs, (short)0, ars, false, false);
        }

        // Parseval-weighted triple product. C/D layout: element v of the
        // accumulator is (M = v + 8*hlf, bin k = tile*16 + m16).
        const int k = tile * 16 + m16;
        float w = (k == 0 || k == DD / 2) ? 1.0f
                 : (k < NBINS ? 2.0f : 0.0f);
        w *= (1.0f / (float)DD);
#pragma unroll
        for (int v = 0; v < 8; ++v) {
            float re = (ahc[v] * atc[v] + ahs[v] * ats[v]) * arc[v]
                     - (ahs[v] * atc[v] - ahc[v] * ats[v]) * ars[v];
            red[v] += w * re;
        }
    }

    // Reduce across the 16 bins held by each 16-lane group (butterfly).
#pragma unroll
    for (int v = 0; v < 8; ++v) {
        float s = red[v];
        s += __shfl_xor(s, 1, 32);
        s += __shfl_xor(s, 2, 32);
        s += __shfl_xor(s, 4, 32);
        s += __shfl_xor(s, 8, 32);
        red[v] = s;
    }
    if (m16 == 0) {
#pragma unroll
        for (int v = 0; v < 8; ++v)
            part[wave][hlf * 8 + v] = red[v];   // M = v + 8*hlf
    }
    __syncthreads();

    // Deterministic fixed-order combine of the 8 wave partials.
    if (tid < ROWS) {
        float s = 0.0f;
#pragma unroll
        for (int w2 = 0; w2 < NTHR / 32; ++w2) s += part[w2][tid];
        out[row0 + tid] = s;
    }
}

// ---------------------------------------------------------------------------
// Launch. setup_inputs order: h, r, t.
// ---------------------------------------------------------------------------
extern "C" void kernel_launch(void* const* d_in, const int* in_sizes, int n_in,
                              void* d_out, int out_size, void* d_ws, size_t ws_size,
                              hipStream_t stream) {
    const float* h = (const float*)d_in[0];
    const float* r = (const float*)d_in[1];
    const float* t = (const float*)d_in[2];
    float* out = (float*)d_out;

    _Float16* cosb = (_Float16*)d_ws;
    _Float16* sinb = cosb + (size_t)NBPAD * DD;

    const int basisN = NBPAD * DD;                 // 139264 f16 elements each
    hole_basis_init<<<(basisN + NTHR - 1) / NTHR, NTHR, 0, stream>>>(cosb, sinb);

    const int nblocks = 131072 / ROWS;             // 8192 workgroups
    hole_main<<<nblocks, NTHR, 0, stream>>>(h, t, r, cosb, sinb, out);
}
